// HGNNLayer_34780645163720
// MI455X (gfx1250) — matrix operand order repaired and checked by reading, not compiled, not run on hardware
//
#include <hip/hip_runtime.h>

#define NN 100000
#define DIM 128

typedef __attribute__((ext_vector_type(16))) _Float16 v16h;
typedef __attribute__((ext_vector_type(8)))  float    v8f;
typedef __attribute__((ext_vector_type(4)))  int      v4i;
typedef __attribute__((address_space(1))) v4i* as1_v4i;
typedef __attribute__((address_space(3))) v4i* as3_v4i;

#if defined(__has_builtin)
#  if __has_builtin(__builtin_amdgcn_global_load_async_to_lds_b128)
#    define HAVE_ASYNC_LDS 1
#  endif
#endif
#ifndef HAVE_ASYNC_LDS
#  define HAVE_ASYNC_LDS 0
#endif

__device__ __forceinline__ void copy16(const _Float16* g, _Float16* l) {
#if HAVE_ASYNC_LDS
  __builtin_amdgcn_global_load_async_to_lds_b128((as1_v4i)(void*)(_Float16*)g,
                                                 (as3_v4i)(void*)l, 0, 0);
#else
  *(uint4*)l = *(const uint4*)g;
#endif
}

__device__ __forceinline__ void staging_fence() {
#if HAVE_ASYNC_LDS
#  if __has_builtin(__builtin_amdgcn_s_wait_asynccnt)
  __builtin_amdgcn_s_wait_asynccnt(0);
#  else
  asm volatile("s_wait_asynccnt 0x0" ::: "memory");
#  endif
#endif
  __syncthreads();
}

// fragment K-offset helpers (per ISA VGPR layout tables, f16 16x16x32)
__device__ __forceinline__ unsigned koA(int j, unsigned ah) {
  return ((j & 4) << 2) + ((j & 3) << 1) + ah;
}
__device__ __forceinline__ unsigned koB(int j, unsigned bh) {
  return ((j & 4) << 2) + ((j & 3) << 1) + (bh >> 1);
}

// ---------------------------------------------------------------- utilities
__global__ void zero_f32(float* p, int n) {
  int i = blockIdx.x * blockDim.x + threadIdx.x;
  if (i < n) p[i] = 0.0f;
}

__global__ void count_kernel(const int* __restrict__ idx1, int groups, int arity,
                             float* __restrict__ cnt) {
  int g = blockIdx.x * blockDim.x + threadIdx.x;
  if (g < groups) unsafeAtomicAdd(&cnt[idx1[g * arity]], 1.0f);
}

// elementwise f32 -> f16
__global__ void conv_h(const float* __restrict__ src, _Float16* __restrict__ dst, int n) {
  int i = blockIdx.x * blockDim.x + threadIdx.x;
  if (i < n) dst[i] = (_Float16)src[i];
}

// A[k][128] (f32) -> At[n][KT] (f16), transposed
__global__ void conv_At(const float* __restrict__ A, _Float16* __restrict__ At, int KT) {
  int i = blockIdx.x * blockDim.x + threadIdx.x;
  if (i < KT * DIM) {
    int k = i >> 7, n = i & 127;
    At[n * KT + k] = (_Float16)A[i];
  }
}

// ------------------------------------------------- dense: out = x @ Cw^T + b
// block: 32 rows x 128 cols, 8 wave32 waves; each wave: one 16-col N-tile,
// two 16-row M-tiles sharing the B fragment.
__global__ void __launch_bounds__(256)
dense_kernel(const _Float16* __restrict__ xh, const _Float16* __restrict__ Cwh,
             const float* __restrict__ Cb, float* __restrict__ out, int nrows) {
  extern __shared__ __align__(16) char smem[];
  const int XS = DIM + 8;                        // padded half stride (16B mult)
  const int CS = DIM + 8;
  _Float16* lx = (_Float16*)smem;                // 32 * XS
  _Float16* lw = lx + 32 * XS;                   // 128 * CS   (Cw rows: [n][k])

  const int t = threadIdx.x;
  const int m0 = blockIdx.x * 32;

  for (int c = t; c < 32 * (DIM / 8); c += 256) {   // 512 chunks of 16B (x)
    int r = c >> 4, q = c & 15;
    int rr = (m0 + r < nrows) ? (m0 + r) : (nrows - 1);
    copy16(xh + rr * DIM + q * 8, lx + r * XS + q * 8);
  }
  for (int c = t; c < DIM * (DIM / 8); c += 256) {  // 2048 chunks (Cw)
    int n = c >> 4, q = c & 15;
    copy16(Cwh + n * DIM + q * 8, lw + n * CS + q * 8);
  }
  staging_fence();

  const int lane = t & 31;
  const int n0   = (t >> 5) * 16;
  const int nl   = lane & 15;
  const unsigned ah = (unsigned)((lane >> 4) << 3);
  const unsigned bh = (unsigned)((lane >> 4) << 4);
  const unsigned* xrow0 = (const unsigned*)(lx + nl * XS);
  const unsigned* xrow1 = (const unsigned*)(lx + (16 + nl) * XS);
  const unsigned* brow  = (const unsigned*)(lw + (n0 + nl) * CS);

  v8f acc0 = {}, acc1 = {};
#pragma unroll
  for (int kk = 0; kk < DIM; kk += 32) {
    union { v16h v; unsigned u[8]; } A0, A1, B;
#pragma unroll
    for (int j = 0; j < 8; ++j) {
      unsigned a = (unsigned)kk + koA(j, ah);
      unsigned b = (unsigned)kk + koB(j, bh);
      A0.u[j] = xrow0[a >> 1];
      A1.u[j] = xrow1[a >> 1];
      B.u[j]  = brow[b >> 1];
    }
    acc0 = __builtin_amdgcn_wmma_f32_16x16x32_f16(false, A0.v, false, B.v,
                                                  (short)0, acc0, false, false);
    acc1 = __builtin_amdgcn_wmma_f32_16x16x32_f16(false, A1.v, false, B.v,
                                                  (short)0, acc1, false, false);
  }

  float bias = Cb[n0 + nl];
  const int mh = (lane >> 4) << 3;
#pragma unroll
  for (int v = 0; v < 8; ++v) {
    int r0 = m0 + v + mh;       r0 = (r0 < nrows) ? r0 : (nrows - 1);
    int r1 = m0 + 16 + v + mh;  r1 = (r1 < nrows) ? r1 : (nrows - 1);
    out[r0 * DIM + n0 + nl] = acc0[v] + bias;   // clamped dups write same value
    out[r1 * DIM + n0 + nl] = acc1[v] + bias;
  }
}

// -------------------------------- messages: msg = gather(x) @ A, scatter-add
// block: 32 groups x 128 cols; wave = one N-tile, two M-tiles sharing B frag.
template <int ARITY>
__global__ void __launch_bounds__(256)
msg_kernel(const _Float16* __restrict__ xh, const int* __restrict__ idx0,
           const int* __restrict__ idx1, const _Float16* __restrict__ Aht,
           const float* __restrict__ cnt, float* __restrict__ out, int groups) {
  extern __shared__ __align__(16) char smem[];
  const int KT = DIM * ARITY;
  const int CR = KT / 8;                         // 16B chunks per row
  const int XS = KT + 8;                         // padded half strides
  const int AS = KT + 8;
  _Float16* lx = (_Float16*)smem;                // 32 * XS   gathered rows
  _Float16* la = lx + 32 * XS;                   // 128 * AS  A^T : [n][k]
  float* lscale = (float*)(la + DIM * AS);       // 32
  int*   ldest  = (int*)(lscale + 32);           // 32

  const int t = threadIdx.x;
  const int g0 = blockIdx.x * 32;

  if (t < 32) {
    int g = g0 + t;
    bool ok = g < groups;
    int gc = ok ? g : (groups - 1);
    int d = idx1[gc * ARITY];
    ldest[t] = d;
    lscale[t] = ok ? (1.0f / cnt[d]) : 0.0f;
  }
  // gathered x rows: 32 rows * CR chunks of 16B
  for (int c = t; c < 32 * CR; c += 256) {
    int r = c / CR;                              // tile row (group slot)
    int q = c - r * CR;                          // 16B chunk within row
    int s = q >> 4;                              // arity slot
    int g = g0 + r;
    int gc = (g < groups) ? g : (groups - 1);
    int node = idx0[gc * ARITY + s];
    copy16(xh + node * DIM + (q & 15) * 8, lx + r * XS + q * 8);
  }
  // A^T tile: 128 rows * CR chunks of 16B
  for (int c = t; c < DIM * CR; c += 256) {
    int n = c / CR;
    int q = c - n * CR;
    copy16(Aht + n * KT + q * 8, la + n * AS + q * 8);
  }
  staging_fence();

  const int lane = t & 31;
  const int n0   = (t >> 5) * 16;
  const int nl   = lane & 15;
  const unsigned ah = (unsigned)((lane >> 4) << 3);
  const unsigned bh = (unsigned)((lane >> 4) << 4);
  const unsigned* xrow0 = (const unsigned*)(lx + nl * XS);
  const unsigned* xrow1 = (const unsigned*)(lx + (16 + nl) * XS);
  const unsigned* brow  = (const unsigned*)(la + (n0 + nl) * AS);

  v8f acc0 = {}, acc1 = {};
#pragma unroll
  for (int kk = 0; kk < KT; kk += 32) {
    union { v16h v; unsigned u[8]; } A0, A1, B;
#pragma unroll
    for (int j = 0; j < 8; ++j) {
      unsigned a = (unsigned)kk + koA(j, ah);
      unsigned b = (unsigned)kk + koB(j, bh);
      A0.u[j] = xrow0[a >> 1];
      A1.u[j] = xrow1[a >> 1];
      B.u[j]  = brow[b >> 1];
    }
    acc0 = __builtin_amdgcn_wmma_f32_16x16x32_f16(false, A0.v, false, B.v,
                                                  (short)0, acc0, false, false);
    acc1 = __builtin_amdgcn_wmma_f32_16x16x32_f16(false, A1.v, false, B.v,
                                                  (short)0, acc1, false, false);
  }

  const int mh = (lane >> 4) << 3;
#pragma unroll
  for (int v = 0; v < 8; ++v) {
    int M = v + mh;
    unsafeAtomicAdd(&out[ldest[M] * DIM + n0 + nl],      acc0[v] * lscale[M]);
    unsafeAtomicAdd(&out[ldest[16 + M] * DIM + n0 + nl], acc1[v] * lscale[16 + M]);
  }
}

// ------------------------------------------------------------------- launch
extern "C" void kernel_launch(void* const* d_in, const int* in_sizes, int n_in,
                              void* d_out, int out_size, void* d_ws, size_t ws_size,
                              hipStream_t stream) {
  const float* x  = (const float*)d_in[0];
  const int*   e1 = (const int*)d_in[1];
  const int*   e2 = (const int*)d_in[2];
  const int*   e3 = (const int*)d_in[3];
  const float* A1 = (const float*)d_in[4];
  const float* A2 = (const float*)d_in[5];
  const float* A3 = (const float*)d_in[6];
  const float* Cw = (const float*)d_in[7];
  const float* Cb = (const float*)d_in[8];
  float* out = (float*)d_out;

  const int E1 = in_sizes[1] / 2, E2 = in_sizes[2] / 2, E3 = in_sizes[3] / 2;
  const int G1 = E1 / 1, G2 = E2 / 2, G3 = E3 / 3;

  // workspace layout
  float* cnt1 = (float*)d_ws;
  float* cnt2 = cnt1 + NN;
  float* cnt3 = cnt2 + NN;
  _Float16* xh   = (_Float16*)(cnt3 + NN);       // NN*128
  _Float16* At1  = xh + (size_t)NN * DIM;        // 128*128
  _Float16* At2  = At1 + DIM * DIM;              // 128*256
  _Float16* At3  = At2 + DIM * 256;              // 128*384
  _Float16* Cwh  = At3 + DIM * 384;              // 128*128

  // 1) degree counts + f16 pre-conversion (independent, stream-ordered)
  zero_f32<<<(3 * NN + 255) / 256, 256, 0, stream>>>(cnt1, 3 * NN);
  count_kernel<<<(G1 + 255) / 256, 256, 0, stream>>>(e1 + E1, G1, 1, cnt1);
  count_kernel<<<(G2 + 255) / 256, 256, 0, stream>>>(e2 + E2, G2, 2, cnt2);
  count_kernel<<<(G3 + 255) / 256, 256, 0, stream>>>(e3 + E3, G3, 3, cnt3);

  conv_h<<<(NN * DIM + 255) / 256, 256, 0, stream>>>(x, xh, NN * DIM);
  conv_At<<<(128 * DIM + 255) / 256, 256, 0, stream>>>(A1, At1, 128);
  conv_At<<<(256 * DIM + 255) / 256, 256, 0, stream>>>(A2, At2, 256);
  conv_At<<<(384 * DIM + 255) / 256, 256, 0, stream>>>(A3, At3, 384);
  conv_h<<<(DIM * DIM + 255) / 256, 256, 0, stream>>>(Cw, Cwh, DIM * DIM);

  // 2) dense projection writes the whole output (no zero-init needed)
  size_t lds_d = 2u * (32 * (DIM + 8) + DIM * (DIM + 8));
  dense_kernel<<<(NN + 31) / 32, 256, lds_d, stream>>>(xh, Cwh, Cb, out, NN);

  // 3) hyperedge messages (stream order gives dense-before-atomics)
  auto ldsz = [](int KT) {
    return (size_t)2u * (32 * (KT + 8) + DIM * (KT + 8)) + 32 * 4 + 32 * 4;
  };
  msg_kernel<1><<<(G1 + 31) / 32, 256, ldsz(128), stream>>>(xh, e1, e1 + E1, At1, cnt1, out, G1);
  msg_kernel<2><<<(G2 + 31) / 32, 256, ldsz(256), stream>>>(xh, e2, e2 + E2, At2, cnt2, out, G2);
  msg_kernel<3><<<(G3 + 31) / 32, 256, ldsz(384), stream>>>(xh, e3, e3 + E3, At3, cnt3, out, G3);
}
